// DecoderLayer_82952998355344
// MI455X (gfx1250) — matrix operand order repaired and checked by reading, not compiled, not run on hardware
//
#include <hip/hip_runtime.h>
#include <hip/hip_bf16.h>
#include <cstdint>

// ---------------------------------------------------------------------------
// Problem dimensions (fixed by the reference)
// ---------------------------------------------------------------------------
constexpr int BATCH = 8;
constexpr int SEQ   = 512;
constexpr int DM    = 512;    // d_model
constexpr int DI    = 1024;   // d_inner
constexpr int DS    = 16;     // d_state
constexpr int DR    = 32;     // dt_rank
constexpr int DF    = 2048;   // d_ff
constexpr int MROWS = BATCH * SEQ;   // 4096 token rows

typedef __attribute__((ext_vector_type(16))) __bf16 v16bf;
typedef __attribute__((ext_vector_type(8)))  float  v8f;

union FragBF { v16bf v; uint4 q[2]; };

// ---------------------------------------------------------------------------
// Utility kernels
// ---------------------------------------------------------------------------
__global__ void cvt_bf16_kernel(const float* __restrict__ in,
                                __hip_bfloat16* __restrict__ out, int n) {
  int i = blockIdx.x * blockDim.x + threadIdx.x;
  if (i < n) out[i] = __float2bfloat16(in[i]);
}

// reverse the sequence dim per batch while converting to bf16
__global__ void flip_cvt_kernel(const float* __restrict__ x,
                                __hip_bfloat16* __restrict__ xr, int n) {
  int i = blockIdx.x * blockDim.x + threadIdx.x;
  if (i >= n) return;
  int c   = i & (DM - 1);
  int row = i >> 9;            // DM == 512
  int b   = row >> 9;          // SEQ == 512
  int l   = row & (SEQ - 1);
  int src = ((b << 9) | (SEQ - 1 - l)) * DM + c;
  xr[i] = __float2bfloat16(x[src]);
}

// ---------------------------------------------------------------------------
// WMMA GEMM:  C[M,N] = A[M,K] * W[N,K]^T (+bias) (+activation)
//   Wave tile 32(M) x 64(N): 2 A-frags x 4 B-frags -> 8 WMMAs per K-step.
//   Block = 8 waves arranged MW x NW.  Software-pipelined (double-buffered
//   register fragments) so next K-step loads are in flight during WMMAs.
//   ACT: 0 = none, 1 = softplus, 2 = exact GELU
// Requires M % (32*MW) == 0, N % (64*NW) == 0, K % 32 == 0 and
// (K == 32 or K % 64 == 0).
// ---------------------------------------------------------------------------
struct Frags { FragBF a[2]; FragBF b[4]; };

__device__ __forceinline__ void load_frags(Frags& F,
                                           const __hip_bfloat16* arow0,
                                           const __hip_bfloat16* arow1,
                                           const __hip_bfloat16* wbase,
                                           int K, int kb, int half) {
  // A fragment (16x32 bf16, ISA 7.12.2): lane half selects K sub-blocks
  const __hip_bfloat16* pa0 = arow0 + kb + half * 8;
  F.a[0].q[0] = *reinterpret_cast<const uint4*>(pa0);
  F.a[0].q[1] = *reinterpret_cast<const uint4*>(pa0 + 16);
  const __hip_bfloat16* pa1 = arow1 + kb + half * 8;
  F.a[1].q[0] = *reinterpret_cast<const uint4*>(pa1);
  F.a[1].q[1] = *reinterpret_cast<const uint4*>(pa1 + 16);
  // B fragments (32x16 bf16): lane holds column N=lane%16, same K sub-blocks as A
#pragma unroll
  for (int t = 0; t < 4; ++t) {
    const __hip_bfloat16* pw = wbase + (size_t)t * 16 * K + kb + half * 8;
    F.b[t].q[0] = *reinterpret_cast<const uint4*>(pw);
    F.b[t].q[1] = *reinterpret_cast<const uint4*>(pw + 16);
  }
}

__device__ __forceinline__ void mma8(v8f acc[2][4], const Frags& F) {
#pragma unroll
  for (int t = 0; t < 4; ++t) {
    acc[0][t] = __builtin_amdgcn_wmma_f32_16x16x32_bf16(
        false, F.a[0].v, false, F.b[t].v, (short)0, acc[0][t], false, false);
    acc[1][t] = __builtin_amdgcn_wmma_f32_16x16x32_bf16(
        false, F.a[1].v, false, F.b[t].v, (short)0, acc[1][t], false, false);
  }
}

template <int ACT, int MW, int NW>
__global__ __launch_bounds__(256)
void wmma_gemm_kernel(const __hip_bfloat16* __restrict__ A,
                      const __hip_bfloat16* __restrict__ W,
                      const float* __restrict__ bias,
                      float* __restrict__ Cf,
                      __hip_bfloat16* __restrict__ Cb,
                      int K, int lda, int ldc) {
  const int lane = threadIdx.x & 31;
  const int wave = threadIdx.x >> 5;
  const int half = lane >> 4;     // 0: lanes 0-15, 1: lanes 16-31
  const int lr   = lane & 15;
  const int mwi  = wave % MW;
  const int nwi  = wave / MW;
  const long m0  = (long)blockIdx.x * (32 * MW) + mwi * 32;
  const long n0  = (long)blockIdx.y * (64 * NW) + nwi * 64;

  v8f acc[2][4] = {};

  const __hip_bfloat16* arow0 = A + (size_t)(m0 + lr) * lda;
  const __hip_bfloat16* arow1 = A + (size_t)(m0 + 16 + lr) * lda;
  const __hip_bfloat16* wbase = W + (size_t)(n0 + lr) * K;

  Frags F0, F1;
  load_frags(F0, arow0, arow1, wbase, K, 0, half);

  int kb = 0;
  for (; kb + 64 <= K; kb += 64) {
    load_frags(F1, arow0, arow1, wbase, K, kb + 32, half);  // in flight
    __builtin_prefetch(arow0 + kb + 128, 0, 1);             // global_prefetch_b8
    mma8(acc, F0);
    if (kb + 96 <= K)
      load_frags(F0, arow0, arow1, wbase, K, kb + 64, half);
    mma8(acc, F1);
  }
  if (kb < K) mma8(acc, F0);   // K == 32 (single-step) path

  // C/D layout: VGPR f, lane l -> M = f + (l/16)*8, N = l%16
#pragma unroll
  for (int t = 0; t < 4; ++t) {
    const long col = n0 + t * 16 + lr;
    const float bv = bias ? bias[col] : 0.f;
#pragma unroll
    for (int i = 0; i < 2; ++i) {
#pragma unroll
      for (int f = 0; f < 8; ++f) {
        long row = m0 + i * 16 + half * 8 + f;
        float v = acc[i][t][f] + bv;
        if (ACT == 1) v = (v > 20.f) ? v : log1pf(__expf(v));               // softplus
        else if (ACT == 2) v = 0.5f * v * (1.f + erff(v * 0.70710678118f)); // gelu
        size_t o = (size_t)row * ldc + col;
        if (Cf) Cf[o] = v;
        if (Cb) Cb[o] = __float2bfloat16(v);
      }
    }
  }
}

// ---------------------------------------------------------------------------
// causal depthwise conv1d (k=4) + bias + SiLU on the "u" half of xz
// ---------------------------------------------------------------------------
__global__ void conv_silu_kernel(const float* __restrict__ xz,
                                 const float* __restrict__ w,     // (DI,4)
                                 const float* __restrict__ cb,    // (DI,)
                                 float* __restrict__ u,
                                 __hip_bfloat16* __restrict__ ubf) {
  int i = blockIdx.x * blockDim.x + threadIdx.x;   // 0 .. MROWS*DI-1
  int c   = i & (DI - 1);
  int row = i >> 10;
  int l   = row & (SEQ - 1);
  float acc = cb[c];
#pragma unroll
  for (int k = 0; k < 4; ++k) {
    int lk = l - 3 + k;
    if (lk >= 0)
      acc += w[c * 4 + k] * xz[(size_t)(row - (3 - k)) * (2 * DI) + c];
  }
  float s = acc / (1.f + __expf(-acc));            // silu
  u[i]   = s;
  ubf[i] = __float2bfloat16(s);
}

// ---------------------------------------------------------------------------
// selective scan: one lane per (batch, d_inner) channel, h[16] in registers,
// fused  y = scan + u*D, gated by silu(z); bf16 output for out-proj GEMM
// ---------------------------------------------------------------------------
__global__ __launch_bounds__(256)
void scan_kernel(const float* __restrict__ delta,  // (MROWS, DI)
                 const float* __restrict__ dbc,    // (MROWS, 64) [dt|B|C]
                 const float* __restrict__ u,      // (MROWS, DI)
                 const float* __restrict__ xz,     // (MROWS, 2*DI), z at +DI
                 const float* __restrict__ A_log,  // (DI, DS)
                 const float* __restrict__ Dp,     // (DI,)
                 __hip_bfloat16* __restrict__ ybf) {
  int t = blockIdx.x * blockDim.x + threadIdx.x;   // 0 .. BATCH*DI-1
  int b = t >> 10;
  int d = t & (DI - 1);
  float a[DS], h[DS];
#pragma unroll
  for (int n = 0; n < DS; ++n) {
    a[n] = -__expf(A_log[d * DS + n]);
    h[n] = 0.f;
  }
  const float Dd = Dp[d];
  for (int l = 0; l < SEQ; ++l) {
    size_t row = (size_t)(b << 9) + l;
    float dl = delta[row * DI + d];
    float ul = u[row * DI + d];
    const float* bc = dbc + row * 64;
    float dlu = dl * ul;
    float y = 0.f;
#pragma unroll
    for (int n = 0; n < DS; ++n) {
      float dA = __expf(dl * a[n]);
      h[n] = dA * h[n] + dlu * bc[DR + n];          // B at cols [32,48)
      y += h[n] * bc[DR + DS + n];                  // C at cols [48,64)
    }
    y += ul * Dd;
    float z  = xz[row * (2 * DI) + DI + d];
    float sz = z / (1.f + __expf(-z));
    ybf[row * DI + d] = __float2bfloat16(y * sz);
  }
}

// ---------------------------------------------------------------------------
// LayerNorm kernels: one wave32 per row of 512 (16 elems/lane)
// ---------------------------------------------------------------------------
__device__ __forceinline__ float wave_sum(float v) {
#pragma unroll
  for (int o = 16; o > 0; o >>= 1) v += __shfl_xor(v, o, 32);
  return v;
}

__global__ __launch_bounds__(256)
void resid_ln1_kernel(const float* __restrict__ x,
                      const float* __restrict__ fwd,
                      const float* __restrict__ bwd,   // in flipped time order
                      const float* __restrict__ g,
                      const float* __restrict__ bb,
                      float* __restrict__ x1,
                      __hip_bfloat16* __restrict__ x1bf) {
  int row  = blockIdx.x * 8 + threadIdx.y;
  int b    = row >> 9;
  int l    = row & (SEQ - 1);
  int rrow = (b << 9) | (SEQ - 1 - l);
  float vals[16];
  float s = 0.f;
#pragma unroll
  for (int i = 0; i < 16; ++i) {
    int c = threadIdx.x + i * 32;
    float v = x[(size_t)row * DM + c] + fwd[(size_t)row * DM + c] +
              bwd[(size_t)rrow * DM + c];
    vals[i] = v;
    s += v;
  }
  float mean = wave_sum(s) * (1.f / DM);
  float s2 = 0.f;
#pragma unroll
  for (int i = 0; i < 16; ++i) {
    float d = vals[i] - mean;
    s2 += d * d;
  }
  float rstd = rsqrtf(wave_sum(s2) * (1.f / DM) + 1e-5f);
#pragma unroll
  for (int i = 0; i < 16; ++i) {
    int c = threadIdx.x + i * 32;
    float o = (vals[i] - mean) * rstd * g[c] + bb[c];
    x1[(size_t)row * DM + c]   = o;
    x1bf[(size_t)row * DM + c] = __float2bfloat16(o);
  }
}

__global__ __launch_bounds__(256)
void resid_ln2_kernel(const float* __restrict__ x1,
                      const float* __restrict__ y2,
                      const float* __restrict__ g,
                      const float* __restrict__ bb,
                      float* __restrict__ out) {
  int row = blockIdx.x * 8 + threadIdx.y;
  float vals[16];
  float s = 0.f;
#pragma unroll
  for (int i = 0; i < 16; ++i) {
    int c = threadIdx.x + i * 32;
    float v = x1[(size_t)row * DM + c] + y2[(size_t)row * DM + c];
    vals[i] = v;
    s += v;
  }
  float mean = wave_sum(s) * (1.f / DM);
  float s2 = 0.f;
#pragma unroll
  for (int i = 0; i < 16; ++i) {
    float d = vals[i] - mean;
    s2 += d * d;
  }
  float rstd = rsqrtf(wave_sum(s2) * (1.f / DM) + 1e-5f);
#pragma unroll
  for (int i = 0; i < 16; ++i) {
    int c = threadIdx.x + i * 32;
    out[(size_t)row * DM + c] = (vals[i] - mean) * rstd * g[c] + bb[c];
  }
}

// ---------------------------------------------------------------------------
// Host side
// ---------------------------------------------------------------------------
static inline void launch_gemm(hipStream_t s, int act,
                               const __hip_bfloat16* A, const __hip_bfloat16* W,
                               const float* bias, float* Cf, __hip_bfloat16* Cb,
                               int M, int N, int K, int lda, int ldc) {
  dim3 b(256);
  if (N % 128 == 0) {                       // MW=4, NW=2: 128x128 block tile
    dim3 g(M / 128, N / 128);
    switch (act) {
      case 0: wmma_gemm_kernel<0, 4, 2><<<g, b, 0, s>>>(A, W, bias, Cf, Cb, K, lda, ldc); break;
      case 1: wmma_gemm_kernel<1, 4, 2><<<g, b, 0, s>>>(A, W, bias, Cf, Cb, K, lda, ldc); break;
      default: wmma_gemm_kernel<2, 4, 2><<<g, b, 0, s>>>(A, W, bias, Cf, Cb, K, lda, ldc); break;
    }
  } else {                                  // MW=8, NW=1: 256x64 block tile
    dim3 g(M / 256, N / 64);
    switch (act) {
      case 0: wmma_gemm_kernel<0, 8, 1><<<g, b, 0, s>>>(A, W, bias, Cf, Cb, K, lda, ldc); break;
      case 1: wmma_gemm_kernel<1, 8, 1><<<g, b, 0, s>>>(A, W, bias, Cf, Cb, K, lda, ldc); break;
      default: wmma_gemm_kernel<2, 8, 1><<<g, b, 0, s>>>(A, W, bias, Cf, Cb, K, lda, ldc); break;
    }
  }
}

static inline void cvt(hipStream_t s, const float* in, __hip_bfloat16* out, int n) {
  cvt_bf16_kernel<<<(n + 255) / 256, 256, 0, s>>>(in, out, n);
}

extern "C" void kernel_launch(void* const* d_in, const int* in_sizes, int n_in,
                              void* d_out, int out_size, void* d_ws, size_t ws_size,
                              hipStream_t stream) {
  (void)in_sizes; (void)n_in; (void)out_size; (void)ws_size;
  const float* x = (const float*)d_in[0];
  // f_* at indices 1..9, r_* at 10..18
  const float* conv1_w = (const float*)d_in[19];
  const float* conv1_b = (const float*)d_in[20];
  const float* conv2_w = (const float*)d_in[21];
  const float* conv2_b = (const float*)d_in[22];
  const float* ln1_g   = (const float*)d_in[23];
  const float* ln1_b   = (const float*)d_in[24];
  const float* ln2_g   = (const float*)d_in[25];
  const float* ln2_b   = (const float*)d_in[26];
  float* out = (float*)d_out;

  // ---- bump-allocate workspace (all sizes are multiples of 256B) ----
  char* ws = (char*)d_ws;
  size_t off = 0;
  auto take = [&](size_t bytes) -> char* {
    char* p = ws + off;
    off += (bytes + 255) & ~(size_t)255;
    return p;
  };
  __hip_bfloat16* XF    = (__hip_bfloat16*)take((size_t)MROWS * DM * 2);
  __hip_bfloat16* XR    = (__hip_bfloat16*)take((size_t)MROWS * DM * 2);
  __hip_bfloat16* W_IN  = (__hip_bfloat16*)take((size_t)2 * DI * DM * 2);
  __hip_bfloat16* W_XP  = (__hip_bfloat16*)take((size_t)(DR + 2 * DS) * DI * 2);
  __hip_bfloat16* W_DT  = (__hip_bfloat16*)take((size_t)DI * DR * 2);
  __hip_bfloat16* W_OUT = (__hip_bfloat16*)take((size_t)DM * DI * 2);
  __hip_bfloat16* W_C1  = (__hip_bfloat16*)take((size_t)DF * DM * 2);
  __hip_bfloat16* W_C2  = (__hip_bfloat16*)take((size_t)DM * DF * 2);
  float*          XZ    = (float*)take((size_t)MROWS * 2 * DI * 4);   // 33.5 MB
  float*          U_F   = (float*)take((size_t)MROWS * DI * 4);       // 16.8 MB
  __hip_bfloat16* U_BF  = (__hip_bfloat16*)take((size_t)MROWS * DI * 2);
  float*          DBC   = (float*)take((size_t)MROWS * 64 * 4);
  __hip_bfloat16* DBCBF = (__hip_bfloat16*)take((size_t)MROWS * 64 * 2);
  float*          DELTA = (float*)take((size_t)MROWS * DI * 4);
  __hip_bfloat16* Y_BF  = (__hip_bfloat16*)take((size_t)MROWS * DI * 2);
  float*          FWD   = (float*)take((size_t)MROWS * DM * 4);
  float*          BWD   = (float*)take((size_t)MROWS * DM * 4);
  // FFN stage aliases over mamba scratch (no longer live then)
  float*          X1    = XZ;
  __hip_bfloat16* X1BF  = (__hip_bfloat16*)((char*)XZ + (size_t)MROWS * DM * 4);
  float*          Y2    = (float*)((char*)X1BF + (size_t)MROWS * DM * 2);
  __hip_bfloat16* H_BF  = (__hip_bfloat16*)U_F;

  // ---- convert x (forward + flipped) and FFN weights to bf16 ----
  cvt(stream, x, XF, MROWS * DM);
  flip_cvt_kernel<<<(MROWS * DM + 255) / 256, 256, 0, stream>>>(x, XR, MROWS * DM);
  cvt(stream, conv1_w, W_C1, DF * DM);
  cvt(stream, conv2_w, W_C2, DM * DF);

  // ---- one mamba direction ----
  auto run_mamba = [&](int base, const __hip_bfloat16* Xbf, float* dir_out) {
    const float* in_w    = (const float*)d_in[base + 0];
    const float* conv_w  = (const float*)d_in[base + 1];
    const float* conv_b  = (const float*)d_in[base + 2];
    const float* xproj_w = (const float*)d_in[base + 3];
    const float* dt_w    = (const float*)d_in[base + 4];
    const float* dt_b    = (const float*)d_in[base + 5];
    const float* A_log   = (const float*)d_in[base + 6];
    const float* Dp      = (const float*)d_in[base + 7];
    const float* out_w   = (const float*)d_in[base + 8];

    cvt(stream, in_w, W_IN, 2 * DI * DM);
    cvt(stream, xproj_w, W_XP, (DR + 2 * DS) * DI);
    cvt(stream, dt_w, W_DT, DI * DR);
    cvt(stream, out_w, W_OUT, DM * DI);

    // xz = X @ in_w^T                               (4096 x 2048)
    launch_gemm(stream, 0, Xbf, W_IN, nullptr, XZ, nullptr,
                MROWS, 2 * DI, DM, DM, 2 * DI);
    // u = silu(causal_conv(u_half) + conv_b)        (4096 x 1024)
    conv_silu_kernel<<<(MROWS * DI + 255) / 256, 256, 0, stream>>>(
        XZ, conv_w, conv_b, U_F, U_BF);
    // dbc = u @ xproj_w^T                           (4096 x 64)
    launch_gemm(stream, 0, U_BF, W_XP, nullptr, DBC, DBCBF,
                MROWS, DR + 2 * DS, DI, DI, DR + 2 * DS);
    // delta = softplus(dt @ dt_w^T + dt_b)          (4096 x 1024)
    launch_gemm(stream, 1, DBCBF, W_DT, dt_b, DELTA, nullptr,
                MROWS, DI, DR, DR + 2 * DS, DI);
    // selective scan + D skip + silu(z) gate -> y (bf16)
    scan_kernel<<<(BATCH * DI) / 256, 256, 0, stream>>>(
        DELTA, DBC, U_F, XZ, A_log, Dp, Y_BF);
    // dir_out = y @ out_w^T                         (4096 x 512)
    launch_gemm(stream, 0, Y_BF, W_OUT, nullptr, dir_out, nullptr,
                MROWS, DM, DI, DI, DM);
  };

  run_mamba(1, XF, FWD);
  run_mamba(10, XR, BWD);

  // x1 = LN1(x + fwd + flip(bwd))
  resid_ln1_kernel<<<MROWS / 8, dim3(32, 8), 0, stream>>>(
      x, FWD, BWD, ln1_g, ln1_b, X1, X1BF);

  // h = gelu(x1 @ conv1_w^T + conv1_b)              (4096 x 2048, bf16)
  launch_gemm(stream, 2, X1BF, W_C1, conv1_b, nullptr, H_BF,
              MROWS, DF, DM, DM, DF);
  // y2 = h @ conv2_w^T + conv2_b                    (4096 x 512)
  launch_gemm(stream, 0, H_BF, W_C2, conv2_b, Y2, nullptr,
              MROWS, DM, DF, DF, DM);
  // out = LN2(x1 + y2)
  resid_ln2_kernel<<<MROWS / 8, dim3(32, 8), 0, stream>>>(
      X1, Y2, ln2_g, ln2_b, out);
}
